// AttentionModel_39848706573401
// MI455X (gfx1250) — matrix operand order repaired and checked by reading, not compiled, not run on hardware
//
#include <hip/hip_runtime.h>

// ---------------------------------------------------------------------------
// Attention: out = softmax((x@Wq+bq) @ (k@Wk+bk)^T / sqrt(E)) @ (v@Wv+bv)
// B=4, S=2048, E=1024.  ~120 GFLOP -> bf16 WMMA (v_wmma_f32_16x16x32_bf16).
// All GEMM B-operands are kept N-major by layout choice (WT transposed at cvt,
// V projection stored transposed), so global->LDS staging is pure contiguous
// GLOBAL_LOAD_ASYNC_TO_LDS_B128 copies with double buffering (ASYNCcnt).
// Workspace (~198 MiB):
//   3x B*S*E bf16 (q/k/v bf16) | 3x E*E bf16 (WqT/WkT/WvT) |
//   Qp,Kp [s][e] bf16, VpT [e][t] bf16 | scores f32 B*S*S | attn bf16 B*S*S
// ---------------------------------------------------------------------------

typedef __attribute__((ext_vector_type(16))) __bf16 v16bf;
typedef __attribute__((ext_vector_type(8)))  __bf16 v8bf;
typedef __attribute__((ext_vector_type(4)))  __bf16 v4bf;
typedef __attribute__((ext_vector_type(8)))  float  v8f;

constexpr int Bb = 4, Ss = 2048, Ee = 1024;

// ------------------------- CDNA5 async-copy helpers ------------------------
__device__ __forceinline__ void async_b128(unsigned lds_addr, unsigned long long gaddr) {
  // per-lane 16B global -> LDS, tracked by ASYNCcnt
  asm volatile("global_load_async_to_lds_b128 %0, %1, off"
               :: "v"(lds_addr), "v"(gaddr) : "memory");
}
template <int N>
__device__ __forceinline__ void wait_async() {
  asm volatile("s_wait_asynccnt %0" :: "n"(N) : "memory");
}

// -------------------------------- fp32 -> bf16 -----------------------------
__global__ __launch_bounds__(256)
void cvt_f32_bf16(const float* __restrict__ s, __bf16* __restrict__ d, int n) {
  int i = (blockIdx.x * 256 + threadIdx.x) * 4;
  int stride = gridDim.x * 256 * 4;
  for (; i < n; i += stride) {
    float4 v = *(const float4*)(s + i);
    v4bf o = { (__bf16)v.x, (__bf16)v.y, (__bf16)v.z, (__bf16)v.w };
    *(v4bf*)(d + i) = o;
  }
}

// fp32 W[in][out] -> bf16 WT[out][in]  (E x E); coalesced writes, L2-served reads
__global__ __launch_bounds__(256)
void cvt_w_transpose(const float* __restrict__ W, __bf16* __restrict__ WT) {
  int idx = (blockIdx.x * 256 + threadIdx.x) * 4;   // over E*E outputs
  int n = idx >> 10;          // out index
  int k = idx & 1023;         // in index (4 consecutive)
  v4bf o = { (__bf16)W[(k + 0) * Ee + n], (__bf16)W[(k + 1) * Ee + n],
             (__bf16)W[(k + 2) * Ee + n], (__bf16)W[(k + 3) * Ee + n] };
  *(v4bf*)(WT + idx) = o;
}

// ------------------------------- bf16 WMMA GEMM ----------------------------
// C[M,N] = scale * (A[M,K] @ B[K,N]) (+ bias[N]);  B stored N-major: Bm[n][k].
// Block tile 128x128, 8 waves (64Mx32N each = 4x2 WMMA tiles), K-step 64,
// double-buffered async global->LDS staging.
template<bool HAS_BIAS, bool C_TRANS, typename CT>
__global__ __launch_bounds__(256, 2)
void gemm_wmma_bf16(const __bf16* __restrict__ A, long bsA,
                    const __bf16* __restrict__ Bm, long bsB,
                    const float* __restrict__ bias,
                    CT* __restrict__ C, long bsC,
                    int K, int lda, int ldb, int ldc, float scale) {
  A  += (long)blockIdx.z * bsA;
  Bm += (long)blockIdx.z * bsB;
  C  += (long)blockIdx.z * bsC;

  constexpr int KSTEP = 64;
  constexpr int LDT   = 72;                    // 144B padded row: conflict-free, 16B aligned
  constexpr unsigned MATB = 128u * LDT * 2u;   // bytes per staged matrix tile
  constexpr unsigned BUFB = 2u * MATB;         // bytes per double-buffer stage
  __shared__ __bf16 Lt[2][2][128 * LDT];       // [buf][A/B][...]

  const int tid  = threadIdx.x;
  const int lane = tid & 31;
  const int w    = tid >> 5, wm = w & 1, wn = w >> 1;
  const int m0   = blockIdx.y * 128, n0 = blockIdx.x * 128;
  const int lh   = lane & 15, kh = lane >> 4;

  // ---- async staging coords: 4 x b128 per matrix per thread ----
  const int sr = tid >> 3;               // rows 0..31, +32 per chunk
  const int sc = (tid & 7) * 8;          // col offset in elements
  const unsigned long long gA = (unsigned long long)(
      reinterpret_cast<unsigned long long>(A) + ((long)(m0 + sr) * lda + sc) * 2);
  const unsigned long long gB = (unsigned long long)(
      reinterpret_cast<unsigned long long>(Bm) + ((long)(n0 + sr) * ldb + sc) * 2);
  const unsigned long long aRow32 = (unsigned long long)lda * 64;  // 32 rows * 2B
  const unsigned long long bRow32 = (unsigned long long)ldb * 64;
  const unsigned ldsA = (unsigned)reinterpret_cast<unsigned long long>(&Lt[0][0][0]) +
                        (unsigned)(sr * LDT + sc) * 2u;
  const unsigned ldsB = ldsA + MATB;

  auto issue = [&](int buf, int kk) {
    const unsigned bo = (unsigned)buf * BUFB;
    const unsigned long long ko = (unsigned long long)kk * 2;
#pragma unroll
    for (int i = 0; i < 4; ++i) {
      async_b128(ldsA + bo + (unsigned)i * (32u * LDT * 2u), gA + ko + i * aRow32);
      async_b128(ldsB + bo + (unsigned)i * (32u * LDT * 2u), gB + ko + i * bRow32);
    }
  };

  const v8f vzero = {};
  v8f acc[4][2];
#pragma unroll
  for (int mt = 0; mt < 4; ++mt)
#pragma unroll
    for (int nt = 0; nt < 2; ++nt) acc[mt][nt] = vzero;

  issue(0, 0);
  for (int kk = 0; kk < K; kk += KSTEP) {
    const int buf = (kk / KSTEP) & 1;
    __syncthreads();                       // all waves done reading buffers
    if (kk + KSTEP < K) {
      issue(buf ^ 1, kk + KSTEP);          // prefetch next stage (8 in flight)
      wait_async<8>();                     // front stage landed (in-order)
    } else {
      wait_async<0>();
    }
    __syncthreads();                       // full tile visible to all waves

    const __bf16* Ab = &Lt[buf][0][0];
    const __bf16* Bt = &Lt[buf][1][0];
#pragma unroll
    for (int ks = 0; ks < KSTEP; ks += 32) {
      v16bf af[4], bfr[2];
      // A 16x32 bf16 frag: lanes 0-15 row=lh K={0..7,16..23}; lanes 16-31 K={8..15,24..31}
#pragma unroll
      for (int mt = 0; mt < 4; ++mt) {
        const __bf16* p = Ab + (wm * 64 + mt * 16 + lh) * LDT + ks + kh * 8;
        v8bf lo = *(const v8bf*)p;
        v8bf hi = *(const v8bf*)(p + 16);
        af[mt] = __builtin_shufflevector(lo, hi, 0, 1, 2, 3, 4, 5, 6, 7,
                                         8, 9, 10, 11, 12, 13, 14, 15);
      }
      // B 32x16 bf16 frag: lane col=lh, lanes 0-15 K={0..7,16..23}; lanes 16-31 K={8..15,24..31}
#pragma unroll
      for (int nt = 0; nt < 2; ++nt) {
        const __bf16* p = Bt + (wn * 32 + nt * 16 + lh) * LDT + ks + kh * 8;
        v8bf lo = *(const v8bf*)p;
        v8bf hi = *(const v8bf*)(p + 16);
        bfr[nt] = __builtin_shufflevector(lo, hi, 0, 1, 2, 3, 4, 5, 6, 7,
                                          8, 9, 10, 11, 12, 13, 14, 15);
      }
#pragma unroll
      for (int mt = 0; mt < 4; ++mt)
#pragma unroll
        for (int nt = 0; nt < 2; ++nt)
          acc[mt][nt] = __builtin_amdgcn_wmma_f32_16x16x32_bf16(
              false, af[mt], false, bfr[nt], (short)0, acc[mt][nt], false, false);
    }
  }

  // ---- epilogue: C VGPR i holds row (i + 8*kh), col lh ----
#pragma unroll
  for (int nt = 0; nt < 2; ++nt) {
    const int col = n0 + wn * 32 + nt * 16 + lh;
    float bv = 0.f;
    if constexpr (HAS_BIAS) bv = bias[col];
#pragma unroll
    for (int mt = 0; mt < 4; ++mt) {
      const int rbase = m0 + wm * 64 + mt * 16 + kh * 8;
      if constexpr (C_TRANS) {             // C[col][row]: 8 rows contiguous -> one b128 store
        v8bf o;
#pragma unroll
        for (int i = 0; i < 8; ++i) o[i] = (__bf16)(acc[mt][nt][i] * scale + bv);
        *(v8bf*)((__bf16*)C + (long)col * ldc + rbase) = o;
      } else {
#pragma unroll
        for (int i = 0; i < 8; ++i) {
          float v = acc[mt][nt][i] * scale + bv;
          C[(long)(rbase + i) * ldc + col] = (CT)v;
        }
      }
    }
  }
}

// --------------------------------- softmax ---------------------------------
// One 256-thread block per row of 2048 f32 scores; writes bf16 attn.
__global__ __launch_bounds__(256)
void softmax_rows(const float* __restrict__ S_, __bf16* __restrict__ Aout) {
  __shared__ float red[256];
  const long row = blockIdx.x;
  const float4* sr4 = (const float4*)(S_ + row * (long)Ss);
  __bf16* ar = Aout + row * (long)Ss;
  const int tid = threadIdx.x;

  float4 x0 = sr4[tid];
  float4 x1 = sr4[tid + 256];
  float v[8] = { x0.x, x0.y, x0.z, x0.w, x1.x, x1.y, x1.z, x1.w };
  float m = -3.0e38f;
#pragma unroll
  for (int j = 0; j < 8; ++j) m = fmaxf(m, v[j]);
  red[tid] = m; __syncthreads();
  for (int s = 128; s > 0; s >>= 1) {
    if (tid < s) red[tid] = fmaxf(red[tid], red[tid + s]);
    __syncthreads();
  }
  m = red[0]; __syncthreads();

  float sum = 0.f;
#pragma unroll
  for (int j = 0; j < 8; ++j) { v[j] = __expf(v[j] - m); sum += v[j]; }
  red[tid] = sum; __syncthreads();
  for (int s = 128; s > 0; s >>= 1) {
    if (tid < s) red[tid] += red[tid + s];
    __syncthreads();
  }
  const float inv = 1.0f / red[0];
  v4bf o0 = { (__bf16)(v[0] * inv), (__bf16)(v[1] * inv),
              (__bf16)(v[2] * inv), (__bf16)(v[3] * inv) };
  v4bf o1 = { (__bf16)(v[4] * inv), (__bf16)(v[5] * inv),
              (__bf16)(v[6] * inv), (__bf16)(v[7] * inv) };
  *(v4bf*)(ar + 4 * tid) = o0;
  *(v4bf*)(ar + 4 * (tid + 256)) = o1;
}

// --------------------------------- launcher --------------------------------
extern "C" void kernel_launch(void* const* d_in, const int* in_sizes, int n_in,
                              void* d_out, int out_size, void* d_ws, size_t ws_size,
                              hipStream_t stream) {
  (void)in_sizes; (void)n_in; (void)out_size; (void)ws_size;
  const float* query = (const float*)d_in[0];
  const float* key_  = (const float*)d_in[1];
  const float* value = (const float*)d_in[2];
  const float* Wq    = (const float*)d_in[3];
  const float* bq    = (const float*)d_in[4];
  const float* Wk    = (const float*)d_in[5];
  const float* bk    = (const float*)d_in[6];
  const float* Wv    = (const float*)d_in[7];
  const float* bv    = (const float*)d_in[8];
  float* out = (float*)d_out;

  char* ws = (char*)d_ws;
  const size_t SZ_IN = (size_t)Bb * Ss * Ee * 2;   // 16 MiB
  const size_t SZ_W  = (size_t)Ee * Ee * 2;        //  2 MiB
  const size_t SZ_SC = (size_t)Bb * Ss * Ss * 4;   // 64 MiB

  __bf16* qb   = (__bf16*)(ws);
  __bf16* kb   = (__bf16*)(ws + SZ_IN);
  __bf16* vb   = (__bf16*)(ws + 2 * SZ_IN);
  __bf16* wqbT = (__bf16*)(ws + 3 * SZ_IN);
  __bf16* wkbT = (__bf16*)(ws + 3 * SZ_IN + SZ_W);
  __bf16* wvbT = (__bf16*)(ws + 3 * SZ_IN + 2 * SZ_W);
  __bf16* Qp   = (__bf16*)(ws + 3 * SZ_IN + 3 * SZ_W);   // [b][s][e]
  __bf16* Kp   = (__bf16*)(ws + 4 * SZ_IN + 3 * SZ_W);   // [b][t][e]
  __bf16* VpT  = (__bf16*)(ws + 5 * SZ_IN + 3 * SZ_W);   // [b][e][t]  (transposed)
  float*  Sc   = (float*) (ws + 6 * SZ_IN + 3 * SZ_W);
  __bf16* Atn  = (__bf16*)(ws + 6 * SZ_IN + 3 * SZ_W + SZ_SC);

  const int nIn = Bb * Ss * Ee, nW = Ee * Ee;

  cvt_f32_bf16<<<nIn / 1024, 256, 0, stream>>>(query, qb, nIn);
  cvt_f32_bf16<<<nIn / 1024, 256, 0, stream>>>(key_,  kb, nIn);
  cvt_f32_bf16<<<nIn / 1024, 256, 0, stream>>>(value, vb, nIn);
  cvt_w_transpose<<<nW / 1024, 256, 0, stream>>>(Wq, wqbT);
  cvt_w_transpose<<<nW / 1024, 256, 0, stream>>>(Wk, wkbT);
  cvt_w_transpose<<<nW / 1024, 256, 0, stream>>>(Wv, wvbT);

  // Q/K projections: [8192,1024] @ WT^T + bias -> bf16 [s][e]
  dim3 gp(Ee / 128, (Bb * Ss) / 128, 1);
  gemm_wmma_bf16<true, false, __bf16><<<gp, 256, 0, stream>>>(
      qb, 0, wqbT, 0, bq, Qp, 0, Ee, Ee, Ee, Ee, 1.0f);
  gemm_wmma_bf16<true, false, __bf16><<<gp, 256, 0, stream>>>(
      kb, 0, wkbT, 0, bk, Kp, 0, Ee, Ee, Ee, Ee, 1.0f);
  // V projection: per-batch, output transposed -> VpT [e][t]
  dim3 gv(Ee / 128, Ss / 128, Bb);
  gemm_wmma_bf16<true, true, __bf16><<<gv, 256, 0, stream>>>(
      vb, (long)Ss * Ee, wvbT, 0, bv, VpT, (long)Ee * Ss, Ee, Ee, Ee, Ss, 1.0f);

  // Scores: Q @ K^T / 32  (Kp is naturally N-major), f32
  dim3 gs(Ss / 128, Ss / 128, Bb);
  gemm_wmma_bf16<false, false, float><<<gs, 256, 0, stream>>>(
      Qp, (long)Ss * Ee, Kp, (long)Ss * Ee, nullptr,
      Sc, (long)Ss * Ss, Ee, Ee, Ee, Ss, 0.03125f);

  // Softmax rows -> bf16 attn
  softmax_rows<<<Bb * Ss, 256, 0, stream>>>(Sc, Atn);

  // Out: attn[s][t] @ V  with B = VpT [e][t] (N-major) -> f32 d_out
  dim3 go(Ee / 128, Ss / 128, Bb);
  gemm_wmma_bf16<false, false, float><<<go, 256, 0, stream>>>(
      Atn, (long)Ss * Ss, VpT, (long)Ee * Ss, nullptr,
      out, (long)Ss * Ee, Ss, Ss, Ss, Ee, 1.0f);
}
